// VarlenMultinomialSampler_35270271434836
// MI455X (gfx1250) — compile-verified, matrix operand order checked by reading, not run on hardware
//
#include <hip/hip_runtime.h>
#include <hip/hip_bf16.h>
#include <math.h>

#define BB   2048
#define DD   256
#define EE   128
#define CCH  64
#define STEPS 10
#define NEGV -1000000.0f

typedef __bf16 bf16;
typedef __attribute__((ext_vector_type(16))) __bf16 v16bf;
typedef __attribute__((ext_vector_type(8)))  float  v8f;

// ---------------- deterministic counter-based RNG ----------------
__device__ __forceinline__ unsigned hash3(unsigned a, unsigned b, unsigned c) {
    unsigned h = a * 0x9E3779B9u ^ b * 0x85EBCA6Bu ^ c * 0xC2B2AE35u;
    h ^= h >> 16; h *= 0x7FEB352Du;
    h ^= h >> 15; h *= 0x846CA68Bu;
    h ^= h >> 16;
    return h;
}
__device__ __forceinline__ float u01(unsigned h) {
    return (float)(h >> 8) * (1.0f / 16777216.0f) + 1e-12f;
}

// ---------------- weight convert: W[K,N] f32 -> Wt[N,K] bf16 ----------------
__global__ void k_w2bf(const float* __restrict__ W, bf16* __restrict__ Wt, int K, int N) {
    int i = blockIdx.x * 256 + threadIdx.x;
    if (i >= K * N) return;
    int n = i / K, k = i % K;
    Wt[i] = (bf16)W[k * N + n];
}

// ---------------- one-time init: mean_pos, sum_val, cur, scalars ----------------
__global__ __launch_bounds__(128) void k_init(
    const int* __restrict__ pos_list, const int* __restrict__ init_samples,
    const float* __restrict__ pos_table, const float* __restrict__ val_embed,
    float* __restrict__ mean_pos, float* __restrict__ sum_val, int* __restrict__ cur,
    float* __restrict__ total_log, int* __restrict__ n_steps, unsigned* __restrict__ active)
{
    int b = blockIdx.x, e = threadIdx.x;
    float ap = 0.f, av = 0.f;
    for (int d = 0; d < DD; ++d) {
        int p = pos_list[b * DD + d];
        int v = init_samples[b * DD + d];
        ap += pos_table[p * EE + e];
        av += val_embed[v * EE + e];
    }
    mean_pos[b * EE + e] = ap * (1.0f / DD);
    sum_val [b * EE + e] = av;
    cur[b * DD + e]        = init_samples[b * DD + e];
    cur[b * DD + e + 128]  = init_samples[b * DD + e + 128];
    if (e == 0) { total_log[b] = 0.f; n_steps[b] = 0; active[b] = 1u; }
}

// ---------------- ctx = mean_pos + sum_val/D -> bf16 (first half of concat buf) ----------------
__global__ void k_ctx(const float* __restrict__ mean_pos, const float* __restrict__ sum_val,
                      bf16* __restrict__ concat)
{
    int i = blockIdx.x * 256 + threadIdx.x;
    if (i >= BB * EE) return;
    int b = i / EE, e = i % EE;
    concat[(long)b * (2 * EE) + e] = (bf16)(mean_pos[i] + sum_val[i] * (1.0f / DD));
}

// ---------------- WMMA GEMM: out[M,N] = relu?(A[M,K] @ Wt[N,K]^T + bias) ----------------
// block = 256 thr = 8 waves; block tile 128(M) x 64(N); wave tile 16 x 64.
// B tile (64 rows x K) is staged once per block into LDS with a padded row
// stride (K+8) so the 16-lane fragment reads spread across all 64 banks.
// Per k-step: issue 4 ds_load B-fragments + next A-fragment global load,
// then 4 back-to-back v_wmma so memory overlaps the matrix pipe.
#define SPAD 8
__global__ __launch_bounds__(256) void k_gemm(
    const bf16* __restrict__ A, int lda,
    const bf16* __restrict__ Wt, const float* __restrict__ bias,
    float* __restrict__ outF, bf16* __restrict__ outB, int ldo, int K, int relu)
{
    __shared__ bf16 sW[64 * (256 + SPAD)];
    const int SK = K + SPAD;

    const int wave  = threadIdx.x >> 5;
    const int lane  = threadIdx.x & 31;
    const int mBase = blockIdx.x * 128 + wave * 16;
    const int nBase = blockIdx.y * 64;
    const int row   = mBase + (lane & 15);
    const int kh    = (lane >> 4) * 16;

    // ---- cooperative stage of the 64 x K weight tile into LDS (8 bf16 chunks) ----
    {
        const int kc8 = K >> 3;                 // chunks per row
        const int nchunks = 64 * kc8;
        for (int c = threadIdx.x; c < nchunks; c += 256) {
            int n  = c / kc8;
            int kc = (c - n * kc8) << 3;
            uint4 v = *(const uint4*)(Wt + (long)(nBase + n) * K + kc);
            *(uint4*)(sW + n * SK + kc) = v;
        }
    }
    __syncthreads();

    v8f acc[4];
    v8f z = {};
#pragma unroll
    for (int t = 0; t < 4; ++t) acc[t] = z;

    const bf16* Arow = A + (long)row * lda + kh;
    v16bf a = *(const v16bf*)(Arow);
    for (int k0 = 0; k0 < K; k0 += 32) {
        // B fragments from LDS (ds_load_b128 x2 each)
        v16bf b0 = *(const v16bf*)(sW + (0 * 16 + (lane & 15)) * SK + k0 + kh);
        v16bf b1 = *(const v16bf*)(sW + (1 * 16 + (lane & 15)) * SK + k0 + kh);
        v16bf b2 = *(const v16bf*)(sW + (2 * 16 + (lane & 15)) * SK + k0 + kh);
        v16bf b3 = *(const v16bf*)(sW + (3 * 16 + (lane & 15)) * SK + k0 + kh);
        // prefetch next A fragment while WMMAs run
        v16bf a_next = a;
        if (k0 + 32 < K) a_next = *(const v16bf*)(Arow + k0 + 32);
        acc[0] = __builtin_amdgcn_wmma_f32_16x16x32_bf16(false, a, false, b0, (short)0, acc[0], false, false);
        acc[1] = __builtin_amdgcn_wmma_f32_16x16x32_bf16(false, a, false, b1, (short)0, acc[1], false, false);
        acc[2] = __builtin_amdgcn_wmma_f32_16x16x32_bf16(false, a, false, b2, (short)0, acc[2], false, false);
        acc[3] = __builtin_amdgcn_wmma_f32_16x16x32_bf16(false, a, false, b3, (short)0, acc[3], false, false);
        a = a_next;
    }

    const int rrow = mBase + (lane >> 4) * 8;
    const int col0 = nBase + (lane & 15);
#pragma unroll
    for (int t = 0; t < 4; ++t) {
        int cc = col0 + t * 16;
        float bcol = bias[cc];
#pragma unroll
        for (int r = 0; r < 8; ++r) {
            float v = acc[t][r] + bcol;
            if (relu) v = fmaxf(v, 0.f);
            long idx = (long)(rrow + r) * ldo + cc;
            if (outF) outF[idx] = v;
            if (outB) outB[idx] = (bf16)v;
        }
    }
}

// ---------------- stop head: dot(h2[b,:256], W2[:,0]) -> sigmoid -> sample ----------------
__global__ __launch_bounds__(256) void k_stop(
    const bf16* __restrict__ act2, const float* __restrict__ W2, const float* __restrict__ b2,
    float* __restrict__ total_log, int* __restrict__ n_steps, unsigned* __restrict__ active, int step)
{
    int wave = threadIdx.x >> 5, lane = threadIdx.x & 31;
    int b = blockIdx.x * 8 + wave;
    const bf16* a = act2 + (long)b * 256 + lane * 8;
    float s = 0.f;
#pragma unroll
    for (int j = 0; j < 8; ++j) s += (float)a[j] * W2[lane * 8 + j];
#pragma unroll
    for (int off = 16; off > 0; off >>= 1) s += __shfl_xor(s, off, 32);
    if (lane == 0) {
        float p = 1.f / (1.f + expf(-(s + b2[0])));
        float u = u01(hash3((unsigned)b, (unsigned)step, 0x5107u));
        int stopped = (u < p);
        unsigned act = active[b];
        float lg = stopped ? logf(p + 1e-18f) : logf(1.f - p + 1e-18f);
        if (act) total_log[b] += lg;
        unsigned still = (act && !stopped) ? 1u : 0u;
        active[b] = still;
        n_steps[b] += (int)still;
    }
}

// ---------------- sample position (softmax + Gumbel argmax over D=256) ----------------
__global__ __launch_bounds__(256) void k_sample_pos(
    const float* __restrict__ pos_logits, const int* __restrict__ cur,
    const float* __restrict__ mod_pos_embed, bf16* __restrict__ concat,
    int* __restrict__ pos_i_buf, int* __restrict__ cur_val_buf,
    float* __restrict__ log_pos_buf, int step)
{
    __shared__ float sV[256];
    __shared__ int   sI[256];
    __shared__ int   sPos;
    int b = blockIdx.x, t = threadIdx.x;
    float l = pos_logits[(long)b * DD + t];

    sV[t] = l; __syncthreads();
    for (int s = 128; s > 0; s >>= 1) { if (t < s) sV[t] = fmaxf(sV[t], sV[t + s]); __syncthreads(); }
    float mx = sV[0]; __syncthreads();
    sV[t] = expf(l - mx); __syncthreads();
    for (int s = 128; s > 0; s >>= 1) { if (t < s) sV[t] += sV[t + s]; __syncthreads(); }
    float lse = mx + logf(sV[0]); __syncthreads();

    float u = u01(hash3((unsigned)b ^ 0xA5A5u, (unsigned)step, (unsigned)t));
    sV[t] = l - logf(-logf(u)); sI[t] = t; __syncthreads();
    for (int s = 128; s > 0; s >>= 1) {
        if (t < s && sV[t + s] > sV[t]) { sV[t] = sV[t + s]; sI[t] = sI[t + s]; }
        __syncthreads();
    }
    if (t == 0) {
        int pi = sI[0];
        sPos = pi;
        pos_i_buf[b]   = pi;
        cur_val_buf[b] = cur[(long)b * DD + pi];
        log_pos_buf[b] = pos_logits[(long)b * DD + pi] - lse;
    }
    __syncthreads();
    int pi = sPos;
    if (t < EE) concat[(long)b * (2 * EE) + EE + t] = (bf16)mod_pos_embed[pi * EE + t];
}

// ---------------- sample value bit + cur/sum_val incremental update ----------------
__global__ __launch_bounds__(128) void k_sample_bit(
    const float* __restrict__ bit_logits, int* __restrict__ cur,
    const int* __restrict__ pos_i_buf, const int* __restrict__ cur_val_buf,
    const float* __restrict__ log_pos_buf, const unsigned* __restrict__ active,
    float* __restrict__ total_log, const float* __restrict__ val_embed,
    float* __restrict__ sum_val, int step)
{
    __shared__ float sV[64];
    __shared__ int   sI[64];
    __shared__ int   sNew, sOld, sChanged;
    int b = blockIdx.x, t = threadIdx.x;
    int cv = cur_val_buf[b];
    float l = NEGV;
    if (t < CCH) {
        l = bit_logits[(long)b * CCH + t];
        if (t == cv) l = NEGV;
        sV[t] = l;
    }
    __syncthreads();
    for (int s = 32; s > 0; s >>= 1) { if (t < s) sV[t] = fmaxf(sV[t], sV[t + s]); __syncthreads(); }
    float mx = sV[0]; __syncthreads();
    if (t < CCH) sV[t] = expf(l - mx);
    __syncthreads();
    for (int s = 32; s > 0; s >>= 1) { if (t < s) sV[t] += sV[t + s]; __syncthreads(); }
    float lse = mx + logf(sV[0]); __syncthreads();
    if (t < CCH) {
        float u = u01(hash3((unsigned)b ^ 0xC3C3u, (unsigned)step, (unsigned)t));
        sV[t] = l - logf(-logf(u)); sI[t] = t;
    }
    __syncthreads();
    for (int s = 32; s > 0; s >>= 1) {
        if (t < s && sV[t + s] > sV[t]) { sV[t] = sV[t + s]; sI[t] = sI[t + s]; }
        __syncthreads();
    }
    if (t == 0) {
        int bi = sI[0];
        float lb = bit_logits[(long)b * CCH + bi];
        if (bi == cv) lb = NEGV;
        float log_bit = lb - lse;
        unsigned still = active[b];
        if (still) total_log[b] += log_pos_buf[b] + log_bit;
        int nv = still ? bi : cv;
        cur[(long)b * DD + pos_i_buf[b]] = nv;
        sNew = nv; sOld = cv; sChanged = (nv != cv) ? 1 : 0;
    }
    __syncthreads();
    if (sChanged)
        sum_val[(long)b * EE + t] += val_embed[sNew * EE + t] - val_embed[sOld * EE + t];
}

// ---------------- finalize outputs ----------------
__global__ void k_finalize(const int* __restrict__ cur, const int* __restrict__ init_samples,
                           const int* __restrict__ n_steps, const float* __restrict__ total_log,
                           float* __restrict__ out)
{
    int i = blockIdx.x * 256 + threadIdx.x;
    int* oi = (int*)out;
    if (i < BB * DD) oi[i] = cur[i];
    if (i < BB) {
        oi[BB * DD + i] = n_steps[i];
        out[BB * DD + BB + i] = total_log[i];
    }
    if (i < BB * DD) oi[BB * DD + 2 * BB + i] = init_samples[i];
}

// ================================================================
extern "C" void kernel_launch(void* const* d_in, const int* in_sizes, int n_in,
                              void* d_out, int out_size, void* d_ws, size_t ws_size,
                              hipStream_t stream)
{
    (void)in_sizes; (void)n_in; (void)out_size; (void)ws_size;

    const int*   pos_list     = (const int*)  d_in[0];
    const int*   init_samples = (const int*)  d_in[1];
    const float* pos_table    = (const float*)d_in[3];
    const float* val_embed    = (const float*)d_in[4];
    const float* mod_pos_emb  = (const float*)d_in[5];
    // MLP params flattened (W,b) pairs in order
    const float* posW[3] = {(const float*)d_in[6],  (const float*)d_in[8],  (const float*)d_in[10]};
    const float* posB[3] = {(const float*)d_in[7],  (const float*)d_in[9],  (const float*)d_in[11]};
    const float* valW[3] = {(const float*)d_in[12], (const float*)d_in[14], (const float*)d_in[16]};
    const float* valB[3] = {(const float*)d_in[13], (const float*)d_in[15], (const float*)d_in[17]};
    const float* stpW[3] = {(const float*)d_in[18], (const float*)d_in[20], (const float*)d_in[22]};
    const float* stpB[3] = {(const float*)d_in[19], (const float*)d_in[21], (const float*)d_in[23]};

    // ---- workspace carve ----
    char* ws = (char*)d_ws;
    size_t off = 0;
    auto carve = [&](size_t bytes) { void* p = ws + off; off = (off + bytes + 255) & ~(size_t)255; return p; };
    bf16* wtS0 = (bf16*)carve(128 * 256 * 2);
    bf16* wtS1 = (bf16*)carve(256 * 256 * 2);
    bf16* wtP0 = (bf16*)carve(128 * 256 * 2);
    bf16* wtP1 = (bf16*)carve(256 * 256 * 2);
    bf16* wtP2 = (bf16*)carve(256 * 256 * 2);
    bf16* wtV0 = (bf16*)carve(256 * 256 * 2);
    bf16* wtV1 = (bf16*)carve(256 * 256 * 2);
    bf16* wtV2 = (bf16*)carve(256 * 64 * 2);
    int*   cur       = (int*)  carve((size_t)BB * DD * 4);
    float* mean_pos  = (float*)carve((size_t)BB * EE * 4);
    float* sum_val   = (float*)carve((size_t)BB * EE * 4);
    bf16*  concat    = (bf16*) carve((size_t)BB * 2 * EE * 2);
    bf16*  act1      = (bf16*) carve((size_t)BB * 256 * 2);
    bf16*  act2      = (bf16*) carve((size_t)BB * 256 * 2);
    float* pos_logit = (float*)carve((size_t)BB * 256 * 4);
    float* bit_logit = (float*)carve((size_t)BB * CCH * 4);
    int*      pos_i   = (int*)     carve((size_t)BB * 4);
    int*      cur_val = (int*)     carve((size_t)BB * 4);
    float*    log_pos = (float*)   carve((size_t)BB * 4);
    float*    tot_log = (float*)   carve((size_t)BB * 4);
    int*      nsteps  = (int*)     carve((size_t)BB * 4);
    unsigned* active  = (unsigned*)carve((size_t)BB * 4);

    auto conv = [&](const float* W, bf16* Wt, int K, int N) {
        int n = K * N;
        k_w2bf<<<(n + 255) / 256, 256, 0, stream>>>(W, Wt, K, N);
    };
    conv(stpW[0], wtS0, 128, 256); conv(stpW[1], wtS1, 256, 256);
    conv(posW[0], wtP0, 128, 256); conv(posW[1], wtP1, 256, 256); conv(posW[2], wtP2, 256, 256);
    conv(valW[0], wtV0, 256, 256); conv(valW[1], wtV1, 256, 256); conv(valW[2], wtV2, 256, 64);

    k_init<<<BB, 128, 0, stream>>>(pos_list, init_samples, pos_table, val_embed,
                                   mean_pos, sum_val, cur, tot_log, nsteps, active);

    auto gemm = [&](const bf16* A, int lda, const bf16* Wt, const float* bias,
                    float* oF, bf16* oB, int ldo, int N, int K, int relu) {
        dim3 g(BB / 128, N / 64);
        k_gemm<<<g, 256, 0, stream>>>(A, lda, Wt, bias, oF, oB, ldo, K, relu);
    };

    for (int step = 0; step < STEPS; ++step) {
        k_ctx<<<(BB * EE + 255) / 256, 256, 0, stream>>>(mean_pos, sum_val, concat);
        // stop MLP: 128 -> 256 -> 256 -> 1
        gemm(concat, 2 * EE, wtS0, stpB[0], nullptr, act1, 256, 256, 128, 1);
        gemm(act1,   256,    wtS1, stpB[1], nullptr, act2, 256, 256, 256, 1);
        k_stop<<<BB / 8, 256, 0, stream>>>(act2, stpW[2], stpB[2], tot_log, nsteps, active, step);
        // pos MLP: 128 -> 256 -> 256 -> 256
        gemm(concat, 2 * EE, wtP0, posB[0], nullptr, act1, 256, 256, 128, 1);
        gemm(act1,   256,    wtP1, posB[1], nullptr, act2, 256, 256, 256, 1);
        gemm(act2,   256,    wtP2, posB[2], pos_logit, nullptr, 256, 256, 256, 0);
        k_sample_pos<<<BB, 256, 0, stream>>>(pos_logit, cur, mod_pos_emb, concat,
                                             pos_i, cur_val, log_pos, step);
        // val MLP: 256 -> 256 -> 256 -> 64
        gemm(concat, 2 * EE, wtV0, valB[0], nullptr, act1, 256, 256, 256, 1);
        gemm(act1,   256,    wtV1, valB[1], nullptr, act2, 256, 256, 256, 1);
        gemm(act2,   256,    wtV2, valB[2], bit_logit, nullptr, 64, 64, 256, 0);
        k_sample_bit<<<BB, 128, 0, stream>>>(bit_logit, cur, pos_i, cur_val, log_pos,
                                             active, tot_log, val_embed, sum_val, step);
    }

    k_finalize<<<(BB * DD + 255) / 256, 256, 0, stream>>>(cur, init_samples, nsteps, tot_log,
                                                          (float*)d_out);
}
